// GraphDestructor_12146167513747
// MI455X (gfx1250) — matrix operand order repaired and checked by reading, not compiled, not run on hardware
//
#include <hip/hip_runtime.h>

#define NN      128
#define NE      2048
#define DIM     128
#define RNDS    2
#define NTYPES  32
#define MROW    272        // 2*DIM + EDGE_DIM
#define EDGEDIM 16

#define LD16    136        // padded f16 leading dim: 272 B row -> 4-bank stagger
#define LD32    132        // padded f32 leading dim

typedef __attribute__((ext_vector_type(16))) _Float16 v16h;
typedef __attribute__((ext_vector_type(8)))  float    v8f;

// ---------------- workspace layout (bytes) ----------------
static constexpr size_t WS_WSRC = 0;                           // [2][128][128] f16
static constexpr size_t WS_WDST = WS_WSRC + (size_t)2*128*128*2;
static constexpr size_t WS_WIH  = WS_WDST + (size_t)2*128*128*2;   // [2][384][128] f16
static constexpr size_t WS_WHH  = WS_WIH  + (size_t)2*384*128*2;
static constexpr size_t WS_HEP  = WS_WHH  + (size_t)2*384*128*2;   // [2][2048][128] f16
static constexpr size_t WS_GUM  = WS_HEP  + (size_t)2*2048*128*2;  // [127][128] f32
static constexpr size_t WS_ROWP = WS_GUM  + 65536;             // 129 ints
static constexpr size_t WS_CSRC = WS_ROWP + 1024;              // 2048 ints
static constexpr size_t WS_CEID = WS_CSRC + 8192;              // 2048 ints
static constexpr size_t WS_VICT = WS_CEID + 8192;              // 127 ints
static constexpr size_t WS_SURV = WS_VICT + 512;               // 1 int

// ---------------- WMMA operand loaders (ISA 7.12.2 layouts) ----------------
// A: 16x32 f16. lane L holds row M = m0+(L&15); halves 0..7 -> K = kb..kb+7,
// halves 8..15 -> K = kb+16..kb+23, kb = (L<16 ? 0 : 8).
__device__ __forceinline__ v16h loadA16(const _Float16* base, int ld, int m0, int k0) {
  int lane = threadIdx.x & 31;
  const _Float16* p = base + (size_t)(m0 + (lane & 15)) * ld + k0 + ((lane & 16) ? 8 : 0);
  union { v16h v; float4 q[2]; } u;
  u.q[0] = *(const float4*)(p);
  u.q[1] = *(const float4*)(p + 16);
  return u.v;
}
// B: 32x16 f16 (B[k][n] = W[n][k], W row-major [out][in]).
// lane L holds col N = n0+(L&15); halves 0..7 -> K = kb..kb+7,
// halves 8..15 -> K = kb+16..kb+23, kb = (L<16 ? 0 : 8).
__device__ __forceinline__ v16h loadB16(const _Float16* base, int ld, int n0, int k0) {
  int lane = threadIdx.x & 31;
  const _Float16* p = base + (size_t)(n0 + (lane & 15)) * ld + k0 + ((lane & 16) ? 8 : 0);
  union { v16h v; float4 q[2]; } u;
  u.q[0] = *(const float4*)(p);
  u.q[1] = *(const float4*)(p + 16);
  return u.v;
}
// C/D: 16x16 f32. lane L, vgpr r -> M = (L<16?0:8)+r, N = L&15.
__device__ __forceinline__ void storeC16(_Float16* base, int ld, int m0, int n0, v8f c) {
  int lane = threadIdx.x & 31;
  int mb = m0 + ((lane & 16) ? 8 : 0);
  int n  = n0 + (lane & 15);
#pragma unroll
  for (int r = 0; r < 8; ++r) base[(size_t)(mb + r) * ld + n] = (_Float16)c[r];
}

#define WMMA_F16(A, B, C) \
  __builtin_amdgcn_wmma_f32_16x16x32_f16(false, (A), false, (B), (short)0, (C), false, false)

// ---------------- prep: cast weights to f16 ----------------
__global__ void k_prep_weights(const float* __restrict__ msgW,
                               const float* __restrict__ gWih,
                               const float* __restrict__ gWhh,
                               _Float16* __restrict__ wsrc, _Float16* __restrict__ wdst,
                               _Float16* __restrict__ wih,  _Float16* __restrict__ whh) {
  const int total = 2*128*128 + 2*128*128 + 2*384*128 + 2*384*128; // 262144
  for (int idx = blockIdx.x * blockDim.x + threadIdx.x; idx < total;
       idx += gridDim.x * blockDim.x) {
    if (idx < 32768) {                    // Wsrc: msg_W[t][o][0..127]
      int t = idx >> 14, rem = idx & 16383, o = rem >> 7, k = rem & 127;
      wsrc[idx] = (_Float16)msgW[(size_t)t * (128*MROW) + o * MROW + k];
    } else if (idx < 65536) {             // Wdst: msg_W[t][o][128..255]
      int j = idx - 32768;
      int t = j >> 14, rem = j & 16383, o = rem >> 7, k = rem & 127;
      wdst[j] = (_Float16)msgW[(size_t)t * (128*MROW) + o * MROW + 128 + k];
    } else if (idx < 65536 + 98304) {     // Wih (same layout)
      int j = idx - 65536;
      wih[j] = (_Float16)gWih[j];
    } else {                               // Whh
      int j = idx - 65536 - 98304;
      whh[j] = (_Float16)gWhh[j];
    }
  }
}

// ---------------- prep: heproj[t][e][o] = he[e] @ Whe[t].T + msg_b[t] ----------------
__global__ void k_heproj(const float* __restrict__ he, const float* __restrict__ msgW,
                         const float* __restrict__ msgb, _Float16* __restrict__ hep) {
  const int total = RNDS * NE * DIM; // 524288
  for (int idx = blockIdx.x * blockDim.x + threadIdx.x; idx < total;
       idx += gridDim.x * blockDim.x) {
    int o = idx & 127, e = (idx >> 7) & (NE - 1), t = idx >> 18;
    float acc = msgb[t * DIM + o];
    const float* wrow = msgW + (size_t)t * (128*MROW) + o * MROW + 256;
    const float* herow = he + (size_t)e * EDGEDIM;
#pragma unroll
    for (int q = 0; q < EDGEDIM; ++q) acc += herow[q] * wrow[q];
    hep[idx] = (_Float16)acc;
  }
}

// ---------------- prep: deterministic CSR by dst ----------------
__global__ void k_csr(const int* __restrict__ src, const int* __restrict__ dst,
                      int* __restrict__ rowptr, int* __restrict__ csrc,
                      int* __restrict__ ceid) {
  __shared__ int cnt[NN];
  int tid = threadIdx.x;
  if (tid < NN) cnt[tid] = 0;
  __syncthreads();
  for (int e = tid; e < NE; e += blockDim.x) atomicAdd(&cnt[dst[e]], 1);
  __syncthreads();
  if (tid == 0) {
    int base = 0;
    rowptr[0] = 0;
    for (int n = 0; n < NN; ++n) { int c = cnt[n]; cnt[n] = base; base += c; rowptr[n + 1] = base; }
    for (int e = 0; e < NE; ++e) {       // sequential -> deterministic order
      int d = dst[e]; int p = cnt[d]++;
      csrc[p] = src[e]; ceid[p] = e;
    }
  }
}

// ---------------- prep: JAX Threefry-2x32 gumbel (key(42) -> (0,42)) ----------------
__device__ __forceinline__ unsigned rotl32(unsigned x, int d) { return (x << d) | (x >> (32 - d)); }
__device__ __forceinline__ float gumbel_from_bits(unsigned b) {
  float f = __uint_as_float(0x3F800000u | (b >> 9)) - 1.0f;   // [0,1)
  const float TINY = 1.17549435e-38f;
  float u = fmaxf(TINY, f * (1.0f - TINY) + TINY);
  return -logf(-logf(u));
}
__global__ void k_gumbel(float* __restrict__ gum) {
  const int HALF = (NN - 1) * NN / 2;  // 8128
  int p = blockIdx.x * blockDim.x + threadIdx.x;
  if (p >= HALF) return;
  unsigned ks0 = 0u, ks1 = 42u, ks2 = ks0 ^ ks1 ^ 0x1BD11BDAu;
  unsigned ks[3] = {ks0, ks1, ks2};
  unsigned x0 = (unsigned)p + ks[0];
  unsigned x1 = (unsigned)(p + HALF) + ks[1];
  const int rot[8] = {13, 15, 26, 6, 17, 29, 16, 24};
#pragma unroll
  for (int i = 0; i < 5; ++i) {
#pragma unroll
    for (int j = 0; j < 4; ++j) {
      int r = rot[(i & 1) * 4 + j];
      x0 += x1; x1 = rotl32(x1, r); x1 ^= x0;
    }
    x0 += ks[(i + 1) % 3];
    x1 += ks[(i + 2) % 3] + (unsigned)(i + 1);
  }
  gum[p]        = gumbel_from_bits(x0);
  gum[p + HALF] = gumbel_from_bits(x1);
}

// ---------------- prep: victim sequence (Gumbel-max over alive) ----------------
__global__ void k_select(const float* __restrict__ gum, int* __restrict__ victims,
                         int* __restrict__ survivor, float* __restrict__ out) {
  __shared__ float sval[NN];
  __shared__ int   sidx[NN];
  __shared__ unsigned salive[4];
  __shared__ int   svict[NN - 1];
  __shared__ float slogp;
  int n = threadIdx.x;
  if (n < 4) salive[n] = 0xFFFFFFFFu;
  if (n == 0) slogp = 0.0f;
  __syncthreads();
  for (int s = 0; s < NN - 1; ++s) {
    bool al = (salive[n >> 5] >> (n & 31)) & 1u;
    sval[n] = al ? gum[s * NN + n] : -1e30f;
    sidx[n] = n;
    __syncthreads();
    for (int off = 64; off > 0; off >>= 1) {
      if (n < off && sval[n + off] > sval[n]) { sval[n] = sval[n + off]; sidx[n] = sidx[n + off]; }
      __syncthreads();
    }
    if (n == 0) {
      int v = sidx[0];
      svict[s] = v;
      victims[s] = v;
      salive[v >> 5] &= ~(1u << (v & 31));
      slogp -= logf((float)(NN - s));        // log p[victim] = -log(count)
    }
    __syncthreads();
  }
  if ((salive[n >> 5] >> (n & 31)) & 1u) { *survivor = n; out[128] = (float)n; }
  if (n < NN - 1) out[129 + n] = (float)svict[NN - 2 - n];   // victims[::-1]
  if (n == 0) out[256] = slogp;
}

// ---------------- persistent GNN scan: one WGP, 32 wave32 waves ----------------
__global__ __launch_bounds__(1024)
void k_gnn(const int* __restrict__ node_types,
           const float* __restrict__ encW, const float* __restrict__ encb,
           const float* __restrict__ gbih, const float* __restrict__ gbhh,
           const _Float16* __restrict__ Wsrc, const _Float16* __restrict__ Wdst,
           const _Float16* __restrict__ Wih,  const _Float16* __restrict__ Whh,
           const _Float16* __restrict__ hep,
           const int* __restrict__ rowptr, const int* __restrict__ csrc,
           const int* __restrict__ ceid,
           const int* __restrict__ victims, const int* __restrict__ survivor,
           float* __restrict__ out) {
  extern __shared__ char smem[];
  _Float16* hv16 = (_Float16*)smem;             // [128][LD16] current hv (f16 mirror)
  _Float16* bufA = hv16 + 128 * LD16;           // Ps, later R
  _Float16* bufB = bufA + 128 * LD16;           // Pd, later Z
  _Float16* a16  = bufB + 128 * LD16;           // aggregated messages
  _Float16* bufC = a16  + 128 * LD16;           // candidate (tanh) output
  float*    hv32 = (float*)(bufC + 128 * LD16); // [128][LD32] f32 master copy
  unsigned* sh_alive = (unsigned*)(hv32 + 128 * LD32);
  int* sh_edges = (int*)(sh_alive + 4);
  int* sh_has   = sh_edges + 1;

  const int tid  = threadIdx.x;
  const int lane = tid & 31;
  const int wv   = tid >> 5;            // wave id 0..31
  const int node = tid >> 3;            // 8 threads per node
  const int col0 = (tid & 7) << 4;      // 16 dims per thread

  // hv0 = one_hot(type) @ enc_W.T + enc_b  ->  enc_W[d][type]
  {
    int ty = node_types[node];
#pragma unroll
    for (int j = 0; j < 16; ++j) {
      float v = encW[(col0 + j) * NTYPES + ty] + encb[col0 + j];
      hv32[node * LD32 + col0 + j] = v;
      hv16[node * LD16 + col0 + j] = (_Float16)v;
    }
  }
  if (tid < 4) sh_alive[tid] = 0xFFFFFFFFu;
  if (tid == 0) { *sh_edges = 0; *sh_has = 0; }
  __syncthreads();

  for (int s = 0; s < NN - 1; ++s) {
    for (int t = 0; t < RNDS; ++t) {
      // ---- Phase 1: Ps = hv@Wsrc.T, Pd = hv@Wdst.T (128 tiles, 4 per wave) ----
#pragma unroll 1
      for (int T = wv; T < 128; T += 32) {
        int outb = T >> 6, tm = (T >> 3) & 7, tn = T & 7;
        const _Float16* W = (outb ? Wdst : Wsrc) + (size_t)t * DIM * DIM;
        v8f acc = {0.f, 0.f, 0.f, 0.f, 0.f, 0.f, 0.f, 0.f};
#pragma unroll 1
        for (int kk = 0; kk < 4; ++kk) {
          v16h A = loadA16(hv16, LD16, tm * 16, kk * 32);
          v16h B = loadB16(W,    DIM,  tn * 16, kk * 32);
          acc = WMMA_F16(A, B, acc);
        }
        storeC16(outb ? bufB : bufA, LD16, tm * 16, tn * 16, acc);
      }
      __syncthreads();

      // ---- Phase 2: a[n] = sum_in(Ps[src]+heproj) + k*Pd[n] ----
      {
        float acc[16];
#pragma unroll
        for (int j = 0; j < 16; ++j) acc[j] = 0.f;
        int e0 = rowptr[node], e1 = rowptr[node + 1];
        int k = 0;
#pragma unroll 1
        for (int e = e0; e < e1; ++e) {
          int sn = csrc[e];
          if ((sh_alive[sn >> 5] >> (sn & 31)) & 1u) {
            ++k;
            const _Float16* ps = bufA + sn * LD16 + col0;
            const _Float16* hp = hep + ((size_t)t * NE + ceid[e]) * DIM + col0;
#pragma unroll
            for (int j = 0; j < 16; ++j) acc[j] += (float)ps[j] + (float)hp[j];
          }
        }
        float kf = (float)k;
        const _Float16* pd = bufB + node * LD16 + col0;
#pragma unroll
        for (int j = 0; j < 16; ++j)
          a16[node * LD16 + col0 + j] = (_Float16)(acc[j] + kf * (float)pd[j]);
        bool nal = (sh_alive[node >> 5] >> (node & 31)) & 1u;
        if (((tid & 7) == 0) && nal && k) atomicAdd(sh_edges, k);
      }
      __syncthreads();
      if (tid == 0) { *sh_has = (*sh_edges > 0) ? 1 : 0; *sh_edges = 0; }
      __syncthreads();

      // ---- Phase 3: r,z gates: sigmoid(a@Wi.T + hv@Wh.T + b) ----
      const float* bih = gbih + t * 384;
      const float* bhh = gbhh + t * 384;
#pragma unroll 1
      for (int T = wv; T < 128; T += 32) {
        int g = T >> 6, tm = (T >> 3) & 7, tn = T & 7;
        const _Float16* Wi = Wih + ((size_t)t * 384 + g * 128) * DIM;
        const _Float16* Wh = Whh + ((size_t)t * 384 + g * 128) * DIM;
        v8f acc = {0.f, 0.f, 0.f, 0.f, 0.f, 0.f, 0.f, 0.f};
#pragma unroll 1
        for (int kk = 0; kk < 4; ++kk)
          acc = WMMA_F16(loadA16(a16, LD16, tm * 16, kk * 32),
                         loadB16(Wi,  DIM,  tn * 16, kk * 32), acc);
#pragma unroll 1
        for (int kk = 0; kk < 4; ++kk)
          acc = WMMA_F16(loadA16(hv16, LD16, tm * 16, kk * 32),
                         loadB16(Wh,   DIM,  tn * 16, kk * 32), acc);
        int colg = g * 128 + tn * 16 + (lane & 15);
        float bv = bih[colg] + bhh[colg];
        _Float16* ob = g ? bufB : bufA;
        int mb = tm * 16 + ((lane & 16) ? 8 : 0);
        int nc = tn * 16 + (lane & 15);
#pragma unroll
        for (int r = 0; r < 8; ++r) {
          float x = acc[r] + bv;
          ob[(mb + r) * LD16 + nc] = (_Float16)(1.0f / (1.0f + __expf(-x)));
        }
      }
      __syncthreads();

      // ---- Phase 4: ncand = tanh((a@Win.T+bi) + r*(hv@Whn.T+bh)) ----
#pragma unroll 1
      for (int T = wv; T < 64; T += 32) {
        int tm = T >> 3, tn = T & 7;
        const _Float16* Wi = Wih + ((size_t)t * 384 + 256) * DIM;
        const _Float16* Wh = Whh + ((size_t)t * 384 + 256) * DIM;
        v8f aih = {0.f, 0.f, 0.f, 0.f, 0.f, 0.f, 0.f, 0.f};
        v8f ahn = {0.f, 0.f, 0.f, 0.f, 0.f, 0.f, 0.f, 0.f};
#pragma unroll 1
        for (int kk = 0; kk < 4; ++kk) {
          aih = WMMA_F16(loadA16(a16,  LD16, tm * 16, kk * 32),
                         loadB16(Wi,   DIM,  tn * 16, kk * 32), aih);
          ahn = WMMA_F16(loadA16(hv16, LD16, tm * 16, kk * 32),
                         loadB16(Wh,   DIM,  tn * 16, kk * 32), ahn);
        }
        int nc = tn * 16 + (lane & 15);
        float bi = bih[256 + nc], bh = bhh[256 + nc];
        int mb = tm * 16 + ((lane & 16) ? 8 : 0);
#pragma unroll
        for (int r = 0; r < 8; ++r) {
          float rr = (float)bufA[(mb + r) * LD16 + nc];
          float cand = tanhf((aih[r] + bi) + rr * (ahn[r] + bh));
          bufC[(mb + r) * LD16 + nc] = (_Float16)cand;
        }
      }
      __syncthreads();

      // ---- Phase 5: hv = upd ? (1-z)*ncand + z*hv : hv ----
      {
        bool upd = (*sh_has != 0) && ((sh_alive[node >> 5] >> (node & 31)) & 1u);
#pragma unroll
        for (int j = 0; j < 16; ++j) {
          float z  = (float)bufB[node * LD16 + col0 + j];
          float cn = (float)bufC[node * LD16 + col0 + j];
          float h  = hv32[node * LD32 + col0 + j];
          if (upd) h = (1.0f - z) * cn + z * h;
          hv32[node * LD32 + col0 + j] = h;
          hv16[node * LD16 + col0 + j] = (_Float16)h;
        }
      }
      __syncthreads();
    } // rounds

    if (tid == 0) {
      int v = victims[s];
      sh_alive[v >> 5] &= ~(1u << (v & 31));
    }
    __syncthreads();
  } // steps

  int surv = *survivor;
  if (node == surv) {
#pragma unroll
    for (int j = 0; j < 16; ++j) out[col0 + j] = hv32[surv * LD32 + col0 + j];
  }
}

// ---------------- launch ----------------
extern "C" void kernel_launch(void* const* d_in, const int* in_sizes, int n_in,
                              void* d_out, int out_size, void* d_ws, size_t ws_size,
                              hipStream_t stream) {
  (void)in_sizes; (void)n_in; (void)out_size; (void)ws_size;
  const int*   node_types = (const int*)d_in[0];
  const int*   src   = (const int*)d_in[1];
  const int*   dst   = (const int*)d_in[2];
  const float* he    = (const float*)d_in[3];
  const float* encW  = (const float*)d_in[4];
  const float* encb  = (const float*)d_in[5];
  const float* msgW  = (const float*)d_in[6];
  const float* msgb  = (const float*)d_in[7];
  const float* gWih  = (const float*)d_in[8];
  const float* gWhh  = (const float*)d_in[9];
  const float* gbih  = (const float*)d_in[10];
  const float* gbhh  = (const float*)d_in[11];
  float* out = (float*)d_out;
  char*  ws  = (char*)d_ws;

  _Float16* wsrc = (_Float16*)(ws + WS_WSRC);
  _Float16* wdst = (_Float16*)(ws + WS_WDST);
  _Float16* wih  = (_Float16*)(ws + WS_WIH);
  _Float16* whh  = (_Float16*)(ws + WS_WHH);
  _Float16* hep  = (_Float16*)(ws + WS_HEP);
  float*    gum  = (float*)(ws + WS_GUM);
  int*      rowp = (int*)(ws + WS_ROWP);
  int*      cs   = (int*)(ws + WS_CSRC);
  int*      ce   = (int*)(ws + WS_CEID);
  int*      vict = (int*)(ws + WS_VICT);
  int*      surv = (int*)(ws + WS_SURV);

  k_prep_weights<<<256, 256, 0, stream>>>(msgW, gWih, gWhh, wsrc, wdst, wih, whh);
  k_heproj<<<512, 256, 0, stream>>>(he, msgW, msgb, hep);
  k_csr<<<1, 256, 0, stream>>>(src, dst, rowp, cs, ce);
  k_gumbel<<<32, 256, 0, stream>>>(gum);
  k_select<<<1, 128, 0, stream>>>(gum, vict, surv, out);

  const size_t smem_bytes = (size_t)5 * 128 * LD16 * sizeof(_Float16)   // 174080
                          + (size_t)128 * LD32 * sizeof(float)          // +67584
                          + 64;                                         // control
  k_gnn<<<1, 1024, smem_bytes, stream>>>(node_types, encW, encb, gbih, gbhh,
                                         wsrc, wdst, wih, whh, hep,
                                         rowp, cs, ce, vict, surv, out);
}
